// AttentionHead_21766894256266
// MI455X (gfx1250) — matrix operand order repaired and checked by reading, not compiled, not run on hardware
//
#include <hip/hip_runtime.h>

typedef __attribute__((ext_vector_type(16))) __bf16 v16bf;
typedef __attribute__((ext_vector_type(8)))  __bf16 v8bf;
typedef __attribute__((ext_vector_type(8)))  float  v8f;
typedef __attribute__((ext_vector_type(4)))  int    v4i;

#define B_       8
#define LQ_      2048
#define LK_      2048
#define DW_      1024
#define DK_      128
#define INV_TEMP 0.08838834764831845f   /* 1/sqrt(128) */
#define NEG_BIG_ (-1.0e9f)

#define XSTR 36    /* fp32 LDS row stride (144 B: 16B-aligned, bank-spread) */
#define WSTR 136   /* bf16 LDS row stride (272 B: 16B-aligned, bank-spread) */
#define OSTR 136

#define XBUF_BYTES 18432   /* 128 * 36 * 4  */
#define WBUF_BYTES 8704    /* 32 * 136 * 2  */
#define WBUF_OFF   36864   /* 2 * XBUF_BYTES */

// -------- CDNA5 async global->LDS copy (ASYNCcnt-tracked), 16B per lane ----
#if __has_builtin(__builtin_amdgcn_global_load_async_to_lds_b128)
#define ASYNC_CP16(g, l)                                                      \
  __builtin_amdgcn_global_load_async_to_lds_b128(                             \
      (__attribute__((address_space(1))) v4i*)(v4i*)(void*)(g),               \
      (__attribute__((address_space(3))) v4i*)(v4i*)(void*)(l), 0, 0)
#else
#define ASYNC_CP16(g, l)                                                      \
  asm volatile("global_load_async_to_lds_b128 %0, %1, off" ::                 \
               "v"((unsigned)(unsigned long long)                             \
                       (__attribute__((address_space(3))) char*)(void*)(l)),  \
               "v"((const void*)(g))                                          \
               : "memory")
#endif

#if __has_builtin(__builtin_amdgcn_s_wait_asynccnt)
#define WAIT_ASYNC(n) __builtin_amdgcn_s_wait_asynccnt(n)
#else
#define WAIT_ASYNC(n) asm volatile("s_wait_asynccnt " #n ::: "memory")
#endif

// ---------------------------------------------------------------------------
// One-time fp32 -> bf16 conversion of a weight matrix (1024x128).
// Grid: 128 blocks x 256 threads, one float4 per thread.
// ---------------------------------------------------------------------------
__global__ __launch_bounds__(256)
void cvt_w_kernel(const float* __restrict__ Wf, __bf16* __restrict__ Wb) {
  int i4 = blockIdx.x * 256 + threadIdx.x;        // 32768 float4 per matrix
  float4 f = ((const float4*)Wf)[i4];
  __bf16* o = Wb + (size_t)i4 * 4;
  o[0] = (__bf16)f.x; o[1] = (__bf16)f.y; o[2] = (__bf16)f.z; o[3] = (__bf16)f.w;
}

// ---------------------------------------------------------------------------
// Projection GEMM: out[row,0:128] = X[row,0:1024] @ W.  X fp32 (async-copied
// to LDS, bf16 convert fused into fragment build), W pre-converted bf16
// (raw async copy).  Double-buffered: tile i+1 transfer overlaps tile i WMMA.
// transpose_out==1 stores ksT[b, d, pos]; epilogue stages through LDS so all
// global stores are coalesced b128.
// Grid: 128 blocks (16384 rows / 128).  Block: 256 threads = 8 waves.
// ---------------------------------------------------------------------------
__global__ __launch_bounds__(256)
void proj_kernel(const float* __restrict__ X, const __bf16* __restrict__ Wb,
                 __bf16* __restrict__ out, int transpose_out) {
  __shared__ __align__(16) char smem[55296];

  const int tid  = threadIdx.x;
  const int wave = tid >> 5;
  const int lane = tid & 31;
  const int m16  = lane & 15;
  const int hi   = lane >> 4;
  const int M0   = blockIdx.x * 128;

  v8f acc[8];
  for (int n = 0; n < 8; ++n) acc[n] = (v8f)0.0f;

  auto issue_tile = [&](int k0, int buf) {
    float*  xf = (float*)(smem + buf * XBUF_BYTES);
    __bf16* wl = (__bf16*)(smem + WBUF_OFF + buf * WBUF_BYTES);
    for (int i = 0; i < 4; ++i) {                 // X tile 128x32 fp32 (16 KB)
      int flat = tid * 16 + i * 4;
      int r = flat >> 5, c = flat & 31;
      ASYNC_CP16(X + (size_t)(M0 + r) * DW_ + k0 + c, xf + r * XSTR + c);
    }
    for (int i = 0; i < 2; ++i) {                 // W tile 32x128 bf16 (8 KB)
      int flat = tid * 16 + i * 8;
      int r = flat >> 7, c = flat & 127;
      ASYNC_CP16(Wb + (size_t)(k0 + r) * DK_ + c, wl + r * WSTR + c);
    }
  };

  issue_tile(0, 0);
  for (int it = 0; it < DW_ / 32; ++it) {
    const int cur = it & 1;
    if (it + 1 < DW_ / 32) {                      // prefetch next tile, then
      issue_tile((it + 1) * 32, cur ^ 1);         // wait only for current one
      WAIT_ASYNC(6);
    } else {
      WAIT_ASYNC(0);
    }
    __syncthreads();

    const float*  xf = (const float*)(smem + cur * XBUF_BYTES);
    const __bf16* wl = (const __bf16*)(smem + WBUF_OFF + cur * WBUF_BYTES);
    v16bf a;                                      // fused fp32->bf16 convert
    for (int e = 0; e < 16; ++e) {
      int k = hi * 8 + (e < 8 ? e : e + 8);
      a[e] = (__bf16)xf[(wave * 16 + m16) * XSTR + k];
    }
    for (int n = 0; n < 8; ++n) {
      v16bf bb;
      for (int e = 0; e < 16; ++e)
        bb[e] = wl[(hi * 8 + (e < 8 ? e : e + 8)) * WSTR + n * 16 + m16];
      acc[n] = __builtin_amdgcn_wmma_f32_16x16x32_bf16(
          false, a, false, bb, (short)0, acc[n], false, false);
    }
    __syncthreads();
  }

  // ---- epilogue: stage tile in LDS, coalesced b128 stores ----
  __bf16* Sout = (__bf16*)smem;                   // reused after K-loop
  for (int n = 0; n < 8; ++n)
    for (int e = 0; e < 8; ++e) {
      int rowl = wave * 16 + hi * 8 + e;
      int col  = n * 16 + m16;
      __bf16 val = (__bf16)acc[n][e];
      if (!transpose_out) Sout[rowl * OSTR + col] = val;
      else                Sout[col * OSTR + rowl] = val;
    }
  __syncthreads();

  const int bsel = M0 >> 11;          // 128-row tile never crosses a batch
  const int pos0 = M0 & (LK_ - 1);
  for (int i = 0; i < 8; ++i) {
    int flat  = tid * 64 + i * 8;     // 16384 bf16, 8 per chunk
    int outer = flat >> 7, inner = flat & 127;
    float4 chunk = *(const float4*)(Sout + outer * OSTR + inner);
    if (!transpose_out)
      *(float4*)(out + (size_t)(M0 + outer) * DK_ + inner) = chunk;
    else
      *(float4*)(out + ((size_t)bsel * DK_ + outer) * LK_ + pos0 + inner) = chunk;
  }
}

// ---------------------------------------------------------------------------
// Flash attention over bf16 projected tensors, fp32 accumulation.
// Grid: (LQ/128, B).  Block: 256 threads = 8 waves; each wave owns 16 q rows.
// ---------------------------------------------------------------------------
__global__ __launch_bounds__(256)
void attn_kernel(const __bf16* __restrict__ qs,   // [B, LQ, 128]
                 const __bf16* __restrict__ ksT,  // [B, 128, LK]
                 const __bf16* __restrict__ vs,   // [B, LK, 128]
                 const int* __restrict__ mlens,
                 float* __restrict__ out) {       // [B, LQ, 128] fp32
  __shared__ __bf16 ldsP[8 * 16 * OSTR];          // per-wave P staging

  const int tid  = threadIdx.x;
  const int wave = tid >> 5;
  const int lane = tid & 31;
  const int m16  = lane & 15;
  const int hi   = lane >> 4;
  const int b    = blockIdx.y;
  const int q0   = blockIdx.x * 128 + wave * 16;
  const int mlen = mlens[b];

  // Q fragments held in registers for all key tiles
  v16bf qa[4];
  const __bf16* qbase = qs + ((size_t)b * LQ_ + q0 + m16) * DK_;
  for (int kk = 0; kk < 4; ++kk) {
    v8bf lo = *(const v8bf*)(qbase + kk * 32 + hi * 8);
    v8bf hv = *(const v8bf*)(qbase + kk * 32 + hi * 8 + 16);
    for (int e = 0; e < 8; ++e) { qa[kk][e] = lo[e]; qa[kk][e + 8] = hv[e]; }
  }

  v8f o[8];
  for (int n = 0; n < 8; ++n) o[n] = (v8f)0.0f;
  float mrow[8], lrow[8];
  for (int e = 0; e < 8; ++e) { mrow[e] = -1.0e30f; lrow[e] = 0.0f; }

  const __bf16* kTb = ksT + (size_t)b * DK_ * LK_;
  const __bf16* vb  = vs  + (size_t)b * LK_ * DK_;

  const int nkt = (mlen + 127) >> 7;
  for (int kt = 0; kt < nkt; ++kt) {
    const int key0 = kt << 7;

    if (kt + 1 < nkt) {                 // prefetch next K/V tile rows
      __builtin_prefetch(kTb + (size_t)(hi * 16 + m16) * LK_ + key0 + 128, 0, 0);
      __builtin_prefetch(vb + (size_t)(key0 + 128 + hi * 16 + m16) * DK_, 0, 0);
    }

    // ---- S = Qs . Ks^T  (128 keys per tile) ----
    v8f s[8];
    for (int n = 0; n < 8; ++n) s[n] = (v8f)0.0f;
    for (int kk = 0; kk < 4; ++kk) {
      for (int n = 0; n < 8; ++n) {
        v16bf bk;
        for (int e = 0; e < 16; ++e)
          bk[e] = kTb[(size_t)(kk * 32 + hi * 8 + (e < 8 ? e : e + 8)) * LK_ + key0 + n * 16 + m16];
        s[n] = __builtin_amdgcn_wmma_f32_16x16x32_bf16(
            false, qa[kk], false, bk, (short)0, s[n], false, false);
      }
    }

    // ---- scale + length mask ----
    float rmax[8];
    for (int e = 0; e < 8; ++e) rmax[e] = -3.0e38f;
    for (int n = 0; n < 8; ++n) {
      int key = key0 + n * 16 + m16;
      for (int e = 0; e < 8; ++e) {
        float v = s[n][e] * INV_TEMP;
        if (key >= mlen) v = NEG_BIG_;
        s[n][e] = v;
        rmax[e] = fmaxf(rmax[e], v);
      }
    }
    for (int off = 1; off < 16; off <<= 1)
      for (int e = 0; e < 8; ++e)
        rmax[e] = fmaxf(rmax[e], __shfl_xor(rmax[e], off, 32));

    // ---- online softmax update ----
    float alpha[8];
    for (int e = 0; e < 8; ++e) {
      float mn = fmaxf(mrow[e], rmax[e]);
      alpha[e] = __expf(mrow[e] - mn);
      mrow[e]  = mn;
    }
    float rsum[8];
    for (int e = 0; e < 8; ++e) rsum[e] = 0.0f;
    for (int n = 0; n < 8; ++n)
      for (int e = 0; e < 8; ++e) {
        float p = __expf(s[n][e] - mrow[e]);
        s[n][e] = p;
        rsum[e] += p;
      }
    for (int off = 1; off < 16; off <<= 1)
      for (int e = 0; e < 8; ++e)
        rsum[e] += __shfl_xor(rsum[e], off, 32);
    for (int e = 0; e < 8; ++e) lrow[e] = lrow[e] * alpha[e] + rsum[e];
    for (int n = 0; n < 8; ++n)
      for (int e = 0; e < 8; ++e) o[n][e] *= alpha[e];

    // ---- stage P (C-layout) to LDS, read back in A-layout (same wave) ----
    __bf16* pw = ldsP + wave * 16 * OSTR;
    for (int n = 0; n < 8; ++n)
      for (int e = 0; e < 8; ++e)
        pw[(hi * 8 + e) * OSTR + n * 16 + m16] = (__bf16)s[n][e];

    v16bf pa[4];
    for (int kk = 0; kk < 4; ++kk)
      for (int e = 0; e < 16; ++e) {
        int k = kk * 32 + hi * 8 + (e < 8 ? e : e + 8);
        pa[kk][e] = pw[m16 * OSTR + k];
      }

    // ---- O += P . Vs ----
    for (int kk = 0; kk < 4; ++kk) {
      for (int n = 0; n < 8; ++n) {
        v16bf bv;
        for (int e = 0; e < 16; ++e)
          bv[e] = vb[(size_t)(key0 + kk * 32 + hi * 8 + (e < 8 ? e : e + 8)) * DK_ + n * 16 + m16];
        o[n] = __builtin_amdgcn_wmma_f32_16x16x32_bf16(
            false, pa[kk], false, bv, (short)0, o[n], false, false);
      }
    }
  }

  // ---- epilogue: 1/l scaling, fp32 store ----
  float inv[8];
  for (int e = 0; e < 8; ++e) inv[e] = 1.0f / lrow[e];
  float* ob = out + ((size_t)b * LQ_ + q0) * DK_;
  for (int n = 0; n < 8; ++n)
    for (int e = 0; e < 8; ++e)
      ob[(size_t)(hi * 8 + e) * DK_ + n * 16 + m16] = o[n][e] * inv[e];
}

// ---------------------------------------------------------------------------
extern "C" void kernel_launch(void* const* d_in, const int* in_sizes, int n_in,
                              void* d_out, int out_size, void* d_ws, size_t ws_size,
                              hipStream_t stream) {
  const float* q  = (const float*)d_in[0];
  const float* k  = (const float*)d_in[1];
  const float* v  = (const float*)d_in[2];
  const int*   ml = (const int*)d_in[3];
  const float* Wq = (const float*)d_in[4];
  const float* Wk = (const float*)d_in[5];
  const float* Wv = (const float*)d_in[6];
  float* out = (float*)d_out;

  // workspace: qs | ksT | vs (4 MB each) | bf16 weights (256 KB each)
  __bf16* qs  = (__bf16*)d_ws;
  __bf16* ksT = qs  + (size_t)B_ * LQ_ * DK_;
  __bf16* vsw = ksT + (size_t)B_ * DK_ * LK_;
  __bf16* wqb = vsw + (size_t)B_ * LK_ * DK_;
  __bf16* wkb = wqb + (size_t)DW_ * DK_;
  __bf16* wvb = wkb + (size_t)DW_ * DK_;

  dim3 blk(256);
  cvt_w_kernel<<<dim3(128), blk, 0, stream>>>(Wq, wqb);
  cvt_w_kernel<<<dim3(128), blk, 0, stream>>>(Wk, wkb);
  cvt_w_kernel<<<dim3(128), blk, 0, stream>>>(Wv, wvb);
  proj_kernel<<<dim3(128), blk, 0, stream>>>(q, wqb, qs, 0);
  proj_kernel<<<dim3(128), blk, 0, stream>>>(k, wkb, ksT, 1);
  proj_kernel<<<dim3(128), blk, 0, stream>>>(v, wvb, vsw, 0);
  attn_kernel<<<dim3(LQ_ / 128, B_), blk, 0, stream>>>(qs, ksT, vsw, ml, out);
}
